// RQ_KMeans_38019050504676
// MI455X (gfx1250) — matrix operand order repaired and checked by reading, not compiled, not run on hardware
//
#include <hip/hip_runtime.h>
#include <hip/hip_bf16.h>

typedef __attribute__((ext_vector_type(16))) __bf16 v16bf;
typedef __attribute__((ext_vector_type(8)))  float  v8f;

#define DD 256
#define KK 1024

// ---------------------------------------------------------------------------
// Prep: codebook f32 -> bf16 hi/lo planes + f32 squared norms.
// One block (256 threads) per codebook entry; grid = L*K.
// ---------------------------------------------------------------------------
__global__ __launch_bounds__(256)
void rq_prep(const float* __restrict__ cb,
             __bf16* __restrict__ cb_hi, __bf16* __restrict__ cb_lo,
             float* __restrict__ cnorm) {
    const int entry = blockIdx.x;           // l*K + k
    const int t = threadIdx.x;              // 0..255 == d
    const size_t off = (size_t)entry * DD + t;
    float v = cb[off];
    __bf16 hi = (__bf16)v;
    float fh = (float)hi;
    __bf16 lo = (__bf16)(v - fh);
    cb_hi[off] = hi;
    cb_lo[off] = lo;

    __shared__ float red[256];
    red[t] = v * v;
    __syncthreads();
    #pragma unroll
    for (int s = 128; s > 0; s >>= 1) {
        if (t < s) red[t] += red[t + s];
        __syncthreads();
    }
    if (t == 0) cnorm[entry] = red[0];
}

// ---------------------------------------------------------------------------
// One RQ layer. Block = 128 threads = 4 waves; each wave owns a 16-row tile.
// Residual is recomputed from x and previously-written ids (layers < layer).
// Score = ||c||^2 - 2*(r . c), dot via bf16 hi/lo split: 3 WMMAs per chunk.
// ---------------------------------------------------------------------------
__global__ __launch_bounds__(128)
void rq_layer(const float* __restrict__ x, const float* __restrict__ cb,
              const __bf16* __restrict__ cb_hi, const __bf16* __restrict__ cb_lo,
              const float* __restrict__ cnorm, int* __restrict__ ids,
              int layer, int L, int N) {
    const int wave = threadIdx.x >> 5;
    const int lane = threadIdx.x & 31;
    const int m = lane & 15;           // row (A) / code (B) / col N (C)
    const int h = lane >> 4;           // lane half
    const int base = blockIdx.x * 64 + wave * 16;
    const int row = base + m;
    const int row_c = (row < N) ? row : (N - 1);

    // --- previous layer assignments for this row ---
    int pidx[4];
    for (int j = 0; j < layer; ++j) pidx[j] = ids[(size_t)row_c * L + j];

    // --- build resident A fragments (hi/lo) for the 16x256 residual tile ---
    // A layout (16-bit, 16x32): lane half h -> elems 0..15 = K d0+16h+e (contiguous)
    v16bf a_hi[8], a_lo[8];
    {
        const float* xr = x + (size_t)row_c * DD;
        #pragma unroll
        for (int c = 0; c < 8; ++c) {
            const int d0 = 32 * c;
            const int s0 = d0 + 16 * h;       // 8 contiguous floats
            const int s1 = d0 + 16 * h + 8;   // 8 contiguous floats
            float tmp[16];
            #pragma unroll
            for (int e = 0; e < 8; ++e) { tmp[e] = xr[s0 + e]; tmp[8 + e] = xr[s1 + e]; }
            for (int j = 0; j < layer; ++j) {
                const float* cr = cb + ((size_t)j * KK + pidx[j]) * DD;
                #pragma unroll
                for (int e = 0; e < 8; ++e) { tmp[e] -= cr[s0 + e]; tmp[8 + e] -= cr[s1 + e]; }
            }
            #pragma unroll
            for (int e = 0; e < 16; ++e) {
                __bf16 hi = (__bf16)tmp[e];
                a_hi[c][e] = hi;
                a_lo[c][e] = (__bf16)(tmp[e] - (float)hi);
            }
        }
    }

    // --- sweep the 1024 codes in 16-wide tiles ---
    float best[8];
    int   bidx[8];
    #pragma unroll
    for (int v = 0; v < 8; ++v) { best[v] = 3.4e38f; bidx[v] = 0; }

    const __bf16* bh_base = cb_hi + (size_t)layer * KK * DD;
    const __bf16* bl_base = cb_lo + (size_t)layer * KK * DD;
    const float*  cn_base = cnorm + (size_t)layer * KK;

    for (int kt = 0; kt < KK / 16; ++kt) {
        const int nIdx = kt * 16 + m;                  // this lane's candidate code
        const __bf16* bh = bh_base + (size_t)nIdx * DD;
        const __bf16* bl = bl_base + (size_t)nIdx * DD;

        v8f acc = {};
        #pragma unroll
        for (int c = 0; c < 8; ++c) {
            // B layout (32x16): lane = N, elems 0..15 = K d0+16h .. +15 (contiguous)
            const int off = 32 * c + 16 * h;
            v16bf bhf = *(const v16bf*)(bh + off);     // 32B aligned
            v16bf blf = *(const v16bf*)(bl + off);
            acc = __builtin_amdgcn_wmma_f32_16x16x32_bf16(
                false, a_hi[c], false, bhf, (short)0, acc, false, false);
            acc = __builtin_amdgcn_wmma_f32_16x16x32_bf16(
                false, a_hi[c], false, blf, (short)0, acc, false, false);
            acc = __builtin_amdgcn_wmma_f32_16x16x32_bf16(
                false, a_lo[c], false, bhf, (short)0, acc, false, false);
        }

        const float cn = cn_base[nIdx];
        #pragma unroll
        for (int v = 0; v < 8; ++v) {
            float s = cn - 2.0f * acc[v];              // ||c||^2 - 2 r.c
            if (s < best[v]) { best[v] = s; bidx[v] = nIdx; }
        }
    }

    // --- cross-lane argmin (16 candidates per row) via LDS ---
    __shared__ float s_best[4][16][16];
    __shared__ int   s_idx [4][16][16];
    #pragma unroll
    for (int v = 0; v < 8; ++v) {
        const int r = v + 8 * h;                       // C layout: M = v + 8h
        s_best[wave][r][m] = best[v];
        s_idx [wave][r][m] = bidx[v];
    }
    __syncthreads();
    if (lane < 16) {
        const int r = lane;
        float bb = s_best[wave][r][0];
        int   bi = s_idx [wave][r][0];
        #pragma unroll
        for (int c2 = 1; c2 < 16; ++c2) {
            float s = s_best[wave][r][c2];
            int   i2 = s_idx[wave][r][c2];
            if (s < bb || (s == bb && i2 < bi)) { bb = s; bi = i2; }
        }
        const int orow = base + r;
        if (orow < N) ids[(size_t)orow * L + layer] = bi;
    }
}

// ---------------------------------------------------------------------------
extern "C" void kernel_launch(void* const* d_in, const int* in_sizes, int n_in,
                              void* d_out, int out_size, void* d_ws, size_t ws_size,
                              hipStream_t stream) {
    const float* x  = (const float*)d_in[0];   // [N, 256]
    const float* cb = (const float*)d_in[1];   // [L, 1024, 256]
    const int N = in_sizes[0] / DD;
    const int L = in_sizes[1] / (KK * DD);

    __bf16* cb_hi = (__bf16*)d_ws;
    __bf16* cb_lo = cb_hi + (size_t)L * KK * DD;
    float*  cnorm = (float*)(cb_lo + (size_t)L * KK * DD);
    int* ids = (int*)d_out;                    // [N, L] int32

    rq_prep<<<L * KK, 256, 0, stream>>>(cb, cb_hi, cb_lo, cnorm);

    const int nblk = (N + 63) / 64;
    for (int l = 0; l < L; ++l) {
        rq_layer<<<nblk, 128, 0, stream>>>(x, cb, cb_hi, cb_lo, cnorm, ids, l, L, N);
    }
}
